// CCN3_12275016532630
// MI455X (gfx1250) — compile-verified, matrix operand order checked by reading, not run on hardware
//
#include <hip/hip_runtime.h>
#include <hip/hip_fp16.h>

typedef __attribute__((ext_vector_type(16))) _Float16 v16h;
typedef __attribute__((ext_vector_type(8)))  _Float16 v8h;
typedef __attribute__((ext_vector_type(8)))  float    v8f;

#define BB 2
#define BN 5000
#define DD 128
#define KK 6

// ---------------------------------------------------------------------------
// Kernel 1: KNN (top-6, stable-argsort semantics) + fused S row computation.
// S[b*N+n, :] = x@W_init + (sum_k (x0[nbr_k] - x))@W_nbr + b_init + 6*b_nbr
// One thread per query point; candidates tiled through LDS.
// ---------------------------------------------------------------------------
__global__ void __launch_bounds__(256)
knn_embed_kernel(const float* __restrict__ loc,
                 const float* __restrict__ deadline,
                 const float* __restrict__ W_init,
                 const float* __restrict__ b_init,
                 const float* __restrict__ W_nbr,
                 const float* __restrict__ b_nbr,
                 _Float16* __restrict__ S)
{
    __shared__ float sx[256];
    __shared__ float sy[256];
    __shared__ float coef[7 * DD];   // rows 0-2: W_init, 3-5: W_nbr, 6: b_init + 6*b_nbr

    const int tid = threadIdx.x;
    const int b   = blockIdx.y;
    const int n   = blockIdx.x * 256 + tid;
    const bool active = (n < BN);

    for (int i = tid; i < 7 * DD; i += 256) {
        int r = i / DD, d = i - r * DD;
        float v;
        if (r < 3)      v = W_init[r * DD + d];
        else if (r < 6) v = W_nbr[(r - 3) * DD + d];
        else            v = b_init[d] + 6.0f * b_nbr[d];
        coef[i] = v;
    }

    float qx = 0.f, qy = 0.f, qd = 0.f;
    if (active) {
        qx = loc[(size_t)(b * BN + n) * 2 + 0];
        qy = loc[(size_t)(b * BN + n) * 2 + 1];
        qd = deadline[b * BN + n];
    }

    float best_d[KK];
    int   best_i[KK];
#pragma unroll
    for (int k = 0; k < KK; ++k) { best_d[k] = 3.4e38f; best_i[k] = 0; }

    for (int tile = 0; tile < BN; tile += 256) {
        int j = tile + tid;
        __syncthreads();
        if (j < BN) {
            sx[tid] = loc[(size_t)(b * BN + j) * 2 + 0];
            sy[tid] = loc[(size_t)(b * BN + j) * 2 + 1];
        }
        __syncthreads();
        int cnt = min(256, BN - tile);
        if (active) {
            for (int t = 0; t < cnt; ++t) {
                float dx = sx[t] - qx;
                float dy = sy[t] - qy;
                float dd = dx * dx + dy * dy;
                if (dd < best_d[KK - 1]) {      // strict <: stable wrt index order
                    best_d[KK - 1] = dd;
                    best_i[KK - 1] = tile + t;
#pragma unroll
                    for (int k = KK - 1; k > 0; --k) {
                        if (best_d[k] < best_d[k - 1]) {
                            float td = best_d[k]; best_d[k] = best_d[k-1]; best_d[k-1] = td;
                            int   ti = best_i[k]; best_i[k] = best_i[k-1]; best_i[k-1] = ti;
                        }
                    }
                }
            }
        }
    }

    if (!active) return;

    // gather neighbours from batch 0 (matches reference's x[0][neighbors])
    float s0 = 0.f, s1 = 0.f, s2 = 0.f;
#pragma unroll
    for (int k = 0; k < KK; ++k) {
        int j = best_i[k];
        s0 += loc[(size_t)j * 2 + 0] - qx;
        s1 += loc[(size_t)j * 2 + 1] - qy;
        s2 += deadline[j] - qd;
    }

    _Float16* srow = S + (size_t)(b * BN + n) * DD;
    for (int d = 0; d < DD; ++d) {
        float v = coef[6 * DD + d]
                + qx * coef[0 * DD + d] + qy * coef[1 * DD + d] + qd * coef[2 * DD + d]
                + s0 * coef[3 * DD + d] + s1 * coef[4 * DD + d] + s2 * coef[5 * DD + d];
        srow[d] = (_Float16)v;
    }
}

// ---------------------------------------------------------------------------
// Kernel 2: W_final (128x128 f32, row-major [k][n]) -> WfT f16 [n][k]
// so the WMMA B tile is a contiguous 16-byte load per lane.
// ---------------------------------------------------------------------------
__global__ void prep_kernel(const float* __restrict__ W_final,
                            _Float16* __restrict__ WfT)
{
    int k = blockIdx.x;     // 0..127
    int d = threadIdx.x;    // 0..127
    WfT[(size_t)d * DD + k] = (_Float16)W_final[(size_t)k * DD + d];
}

// ---------------------------------------------------------------------------
// Kernel 3: depot embedding -> h[b, 0, :]
// ---------------------------------------------------------------------------
__global__ void depot_kernel(const float* __restrict__ depot,
                             const float* __restrict__ W_depot,
                             const float* __restrict__ b_depot,
                             float* __restrict__ h)
{
    int b = blockIdx.x;
    int d = threadIdx.x;
    float v = b_depot[d]
            + depot[b * 2 + 0] * W_depot[0 * DD + d]
            + depot[b * 2 + 1] * W_depot[1 * DD + d];
    h[(size_t)b * (BN + 1) * DD + d] = v;
}

// ---------------------------------------------------------------------------
// Kernel 4: F_final = S(10000x128) @ W_final(128x128) + 7*b_final  via WMMA.
// Block = 256 threads = 8 waves; each wave owns one 16x16 output tile
// (block row-strip of 16 rows, wave w -> columns 16w..16w+15). K unrolled 4x.
// A layout per ISA: lane L holds row L%16; halves a[0..7]=K(half*8+i),
// a[8..15]=K(16+half*8+i). B identical with columns (WfT is column-major).
// D layout: lane L holds col L%16, VGPR r holds row 8*(L/16)+r.
// ---------------------------------------------------------------------------
__global__ void __launch_bounds__(256)
wmma_final_kernel(const _Float16* __restrict__ S,
                  const _Float16* __restrict__ WfT,
                  const float* __restrict__ b_final,
                  float* __restrict__ h)
{
    const int lane = threadIdx.x & 31;
    const int wave = threadIdx.x >> 5;
    const int row0 = blockIdx.x * 16;
    const int col0 = wave * 16;
    const int lmod = lane & 15;
    const int half = lane >> 4;

    const _Float16* arow = S   + (size_t)(row0 + lmod) * DD + half * 8;
    const _Float16* bcol = WfT + (size_t)(col0 + lmod) * DD + half * 8;

    v8f acc = {};
#pragma unroll
    for (int kb = 0; kb < DD; kb += 32) {
        v8h alo = *(const v8h*)(arow + kb);
        v8h ahi = *(const v8h*)(arow + kb + 16);
        v8h blo = *(const v8h*)(bcol + kb);
        v8h bhi = *(const v8h*)(bcol + kb + 16);
        v16h a  = __builtin_shufflevector(alo, ahi, 0,1,2,3,4,5,6,7,8,9,10,11,12,13,14,15);
        v16h bm = __builtin_shufflevector(blo, bhi, 0,1,2,3,4,5,6,7,8,9,10,11,12,13,14,15);
        acc = __builtin_amdgcn_wmma_f32_16x16x32_f16(
                /*neg_a=*/false, a, /*neg_b=*/false, bm,
                /*c_mod=*/(short)0, acc, /*reuse_a=*/false, /*reuse_b=*/false);
    }

    const float bias = 7.0f * b_final[col0 + lmod];   // sum over 7 concat rows
#pragma unroll
    for (int r = 0; r < 8; ++r) {
        int g = row0 + half * 8 + r;                  // global row in [0, B*N)
        int b = (g >= BN) ? 1 : 0;                    // cheap split: no integer div
        int n = g - b * BN;
        h[((size_t)b * (BN + 1) + 1 + n) * DD + col0 + lmod] = acc[r] + bias;
    }
}

// ---------------------------------------------------------------------------
// Kernel 5: mean over axis=1 of h (B, N+1, D)
// ---------------------------------------------------------------------------
__global__ void mean_kernel(const float* __restrict__ h,
                            float* __restrict__ mean_out)
{
    int b = blockIdx.x;
    int d = threadIdx.x;
    const float* base = h + (size_t)b * (BN + 1) * DD + d;
    float s = 0.f;
    for (int i = 0; i <= BN; ++i) s += base[(size_t)i * DD];
    mean_out[b * DD + d] = s * (1.0f / (float)(BN + 1));
}

extern "C" void kernel_launch(void* const* d_in, const int* in_sizes, int n_in,
                              void* d_out, int out_size, void* d_ws, size_t ws_size,
                              hipStream_t stream)
{
    const float* loc      = (const float*)d_in[0];
    const float* deadline = (const float*)d_in[1];
    const float* depot    = (const float*)d_in[2];
    const float* W_init   = (const float*)d_in[3];
    const float* b_init   = (const float*)d_in[4];
    const float* W_nbr    = (const float*)d_in[5];
    const float* b_nbr    = (const float*)d_in[6];
    const float* W_depot  = (const float*)d_in[7];
    const float* b_depot  = (const float*)d_in[8];
    const float* W_final  = (const float*)d_in[9];
    const float* b_final  = (const float*)d_in[10];

    float* h        = (float*)d_out;                         // (B, N+1, D)
    float* mean_out = h + (size_t)BB * (BN + 1) * DD;        // (B, D)

    _Float16* S   = (_Float16*)d_ws;                         // (B*N, D) f16
    _Float16* WfT = S + (size_t)BB * BN * DD;                // (D, D) f16, [n][k]

    knn_embed_kernel<<<dim3((BN + 255) / 256, BB), 256, 0, stream>>>(
        loc, deadline, W_init, b_init, W_nbr, b_nbr, S);
    prep_kernel<<<DD, DD, 0, stream>>>(W_final, WfT);
    depot_kernel<<<BB, DD, 0, stream>>>(depot, W_depot, b_depot, h);
    wmma_final_kernel<<<(BB * BN) / 16, 256, 0, stream>>>(S, WfT, b_final, h);
    mean_kernel<<<BB, DD, 0, stream>>>(h, mean_out);
}
